// MultiHeadAttention_9594956939709
// MI455X (gfx1250) — matrix operand order repaired and checked by reading, not compiled, not run on hardware
//
#include <hip/hip_runtime.h>

// ---------------------------------------------------------------------------
// Multi-head attention for MI455X (gfx1250, wave32, WMMA).
//   B=4, L=2048, E=1024, H=16, D=64
// Pipeline:
//   1) convert x -> bf16; build fused [Wq|Wk|Wv] bf16 (1024x3072); Wo -> bf16
//   2) QKV = x @ [Wq|Wk|Wv]   (one bf16 WMMA GEMM, N=3072, bf16 out)
//   3) flash attention per (b,h,64-query-block), bf16 out
//   4) out = attn @ Wo + bo   (bf16 WMMA GEMM, fp32 out)
// ---------------------------------------------------------------------------

typedef __attribute__((ext_vector_type(16))) __bf16 v16bf;
typedef __attribute__((ext_vector_type(8)))  float  v8f;
typedef __attribute__((ext_vector_type(4)))  int    v4i;

union AF { v16bf v; uint4 q[2]; };                 // 16x32 bf16 A-fragment
union BF { v16bf v; uint4 q[2]; uint2 d[4]; };     // 32x16 bf16 B-fragment
union CF { v8f  v; float f[8]; };                  // 16x16 f32  C/D fragment

#if __has_builtin(__builtin_amdgcn_global_load_async_to_lds_b128) && \
    __has_builtin(__builtin_amdgcn_s_wait_asynccnt)
#define USE_ASYNC_LDS 1
#else
#define USE_ASYNC_LDS 0
#endif

#if __has_builtin(__builtin_amdgcn_update_dpp)
#define USE_DPP 1
#else
#define USE_DPP 0
#endif

static __device__ __forceinline__ unsigned short f2bf(float x) {
  union { float f; unsigned int u; } c; c.f = x;
  unsigned int u = c.u;
  return (unsigned short)((u + 0x7FFFu + ((u >> 16) & 1u)) >> 16);  // RNE
}

// async global->LDS 16B copy, tracked on ASYNCcnt
// prototype (from compiler diagnostic): (v4i AS1*, v4i AS3*, imm offset, imm cpol)
static __device__ __forceinline__ void async_b128(const unsigned short* g,
                                                  unsigned short* l) {
#if USE_ASYNC_LDS
  __builtin_amdgcn_global_load_async_to_lds_b128(
      (__attribute__((address_space(1))) v4i*)(g),
      (__attribute__((address_space(3))) v4i*)(l), 0, 0);
#endif
}
static __device__ __forceinline__ void wait_async0() {
#if USE_ASYNC_LDS
  __builtin_amdgcn_s_wait_asynccnt(0);
#endif
}

// 16-lane butterfly reductions (stay within each 16-lane half of the wave).
#if USE_DPP
#define DPP_MAX(x, ctrl)                                                     \
  x = fmaxf(x, __builtin_bit_cast(float, __builtin_amdgcn_update_dpp(        \
                   0, __builtin_bit_cast(int, x), ctrl, 0xF, 0xF, true)))
#define DPP_ADD(x, ctrl)                                                     \
  x = x + __builtin_bit_cast(float, __builtin_amdgcn_update_dpp(             \
              0, __builtin_bit_cast(int, x), ctrl, 0xF, 0xF, true))
static __device__ __forceinline__ float red_max16(float x) {
  DPP_MAX(x, 0xB1);   // quad_perm(1,0,3,2)  : xor 1
  DPP_MAX(x, 0x4E);   // quad_perm(2,3,0,1)  : xor 2
  DPP_MAX(x, 0x141);  // row_half_mirror     : combine quads within 8
  DPP_MAX(x, 0x140);  // row_mirror          : combine halves within 16
  return x;
}
static __device__ __forceinline__ float red_sum16(float x) {
  DPP_ADD(x, 0xB1);
  DPP_ADD(x, 0x4E);
  DPP_ADD(x, 0x141);
  DPP_ADD(x, 0x140);
  return x;
}
#else
static __device__ __forceinline__ float red_max16(float x) {
#pragma unroll
  for (int off = 1; off < 16; off <<= 1) x = fmaxf(x, __shfl_xor(x, off, 16));
  return x;
}
static __device__ __forceinline__ float red_sum16(float x) {
#pragma unroll
  for (int off = 1; off < 16; off <<= 1) x += __shfl_xor(x, off, 16);
  return x;
}
#endif

// ---------------------------------------------------------------------------
// fp32 -> bf16, vectorized x4 (n divisible by 4)
// ---------------------------------------------------------------------------
__global__ void cvt4_f32_bf16(const float* __restrict__ in,
                              unsigned short* __restrict__ out, int n4) {
  int i = blockIdx.x * blockDim.x + threadIdx.x;
  int stride = gridDim.x * blockDim.x;
  for (; i < n4; i += stride) {
    float4 f = ((const float4*)in)[i];
    ushort4 o;
    o.x = f2bf(f.x); o.y = f2bf(f.y); o.z = f2bf(f.z); o.w = f2bf(f.w);
    ((ushort4*)out)[i] = o;
  }
}

// fp32 E x E -> bf16 into a 1024 x 3072 concatenated buffer at column offset
__global__ void cvt4_w_cat(const float* __restrict__ in,
                           unsigned short* __restrict__ out, int n4) {
  int i = blockIdx.x * blockDim.x + threadIdx.x;
  int stride = gridDim.x * blockDim.x;
  for (; i < n4; i += stride) {
    float4 f = ((const float4*)in)[i];
    ushort4 o;
    o.x = f2bf(f.x); o.y = f2bf(f.y); o.z = f2bf(f.z); o.w = f2bf(f.w);
    const int i4 = i * 4;
    const int r = i4 >> 10;        // row in 1024-wide source
    const int c = i4 & 1023;       // col
    *(ushort4*)(out + (size_t)r * 3072 + c) = o;
  }
}

// ---------------------------------------------------------------------------
// bf16 WMMA GEMM: C[M,N] = A[M,K] @ B[K,N]  (row-major, strides lda/ldb/ldc)
// Tile: 128x128x64 per 256-thread workgroup (8 waves), 64x32 per wave.
// OUT_F32=0: bf16 output, no bias.  OUT_F32=1: f32 output + bias.
// ---------------------------------------------------------------------------
template <int OUT_F32>
__global__ __launch_bounds__(256)
void gemm_bf16(const unsigned short* __restrict__ A, int lda,
               const unsigned short* __restrict__ B, int ldb,
               void* __restrict__ Cout, int ldc,
               const float* __restrict__ bias, int K) {
  __shared__ __align__(16) unsigned short As[128 * 64];  // row-major, stride 64
  __shared__ __align__(16) unsigned short Bt[128 * 72];  // n-major (transposed), stride 72

  const int tid  = threadIdx.x;
  const int w    = tid >> 5;
  const int lane = tid & 31;
  const int half = lane >> 4;
  const int ln   = lane & 15;
  const int m0 = blockIdx.y * 128;
  const int n0 = blockIdx.x * 128;
  const int wm = (w >> 2) * 64;  // wave row offset
  const int wn = (w & 3) * 32;   // wave col offset

  CF acc[4][2];
#pragma unroll
  for (int i = 0; i < 4; i++)
#pragma unroll
    for (int j = 0; j < 2; j++)
#pragma unroll
      for (int e = 0; e < 8; e++) acc[i][j].f[e] = 0.f;

  // cooperative-load indices: 32 elements (64B) per thread
  const int ar  = tid >> 1;        // A row 0..127
  const int as  = (tid & 1) * 32;  // A col segment
  const int bkr = tid >> 2;        // B k-row 0..63
  const int bns = (tid & 3) * 32;  // B n segment

  for (int k0 = 0; k0 < K; k0 += 64) {
    // B tile -> registers (issue before the barrier for overlap)
    const unsigned short* gb = B + (size_t)(k0 + bkr) * ldb + n0 + bns;
    union { uint4 q[4]; unsigned short s[32]; } bu;
#pragma unroll
    for (int i = 0; i < 4; i++) bu.q[i] = ((const uint4*)gb)[i];
#if !USE_ASYNC_LDS
    const unsigned short* gaf = A + (size_t)(m0 + ar) * lda + k0 + as;
    uint4 areg[4];
#pragma unroll
    for (int i = 0; i < 4; i++) areg[i] = ((const uint4*)gaf)[i];
#endif

    __syncthreads();  // previous iteration's LDS reads complete

#if USE_ASYNC_LDS
    {
      const unsigned short* ga = A + (size_t)(m0 + ar) * lda + k0 + as;
      unsigned short* la = &As[ar * 64 + as];
#pragma unroll
      for (int i = 0; i < 4; i++) async_b128(ga + 8 * i, la + 8 * i);
    }
#else
#pragma unroll
    for (int i = 0; i < 4; i++) *(uint4*)(&As[ar * 64 + as + 8 * i]) = areg[i];
#endif
    // transpose-scatter B into n-major LDS
#pragma unroll
    for (int e = 0; e < 32; e++) Bt[(bns + e) * 72 + bkr] = bu.s[e];

    wait_async0();
    __syncthreads();

#pragma unroll
    for (int ks = 0; ks < 2; ks++) {
      AF af[4];
#pragma unroll
      for (int mt = 0; mt < 4; mt++) {
        const unsigned short* p = &As[(wm + mt * 16 + ln) * 64 + ks * 32 + half * 8];
        af[mt].q[0] = *(const uint4*)(p);
        af[mt].q[1] = *(const uint4*)(p + 16);
      }
      BF bfr[2];
#pragma unroll
      for (int nt = 0; nt < 2; nt++) {
        const unsigned short* p = &Bt[(wn + nt * 16 + ln) * 72 + ks * 32 + half * 8];
        bfr[nt].q[0] = *(const uint4*)(p);      // 144B row stride -> 16B aligned
        bfr[nt].q[1] = *(const uint4*)(p + 16);
      }
#pragma unroll
      for (int mt = 0; mt < 4; mt++)
#pragma unroll
        for (int nt = 0; nt < 2; nt++)
          acc[mt][nt].v = __builtin_amdgcn_wmma_f32_16x16x32_bf16(
              false, af[mt].v, false, bfr[nt].v, (short)0, acc[mt][nt].v, false, false);
    }
  }

  // epilogue: C layout = VGPR r -> row (r + 8*half), col = ln
#pragma unroll
  for (int mt = 0; mt < 4; mt++) {
#pragma unroll
    for (int nt = 0; nt < 2; nt++) {
      const int col = n0 + wn + nt * 16 + ln;
      float bv = 0.f;
      if (OUT_F32) bv = bias[col];
#pragma unroll
      for (int r = 0; r < 8; r++) {
        const int row = m0 + wm + mt * 16 + r + half * 8;
        if (OUT_F32) {
          ((float*)Cout)[(size_t)row * ldc + col] = acc[mt][nt].f[r] + bv;
        } else {
          ((unsigned short*)Cout)[(size_t)row * ldc + col] = f2bf(acc[mt][nt].f[r]);
        }
      }
    }
  }
}

// ---------------------------------------------------------------------------
// Flash attention: grid (L/64, H, B), 128 threads (4 waves).
// Q/K/V live in the fused QKV buffer with row stride ldq (=3072).
// scale = 1/sqrt(E) = 1/32 (reference scales by sqrt(embed_size)).
// ---------------------------------------------------------------------------
__global__ __launch_bounds__(128)
void attn_flash(const unsigned short* __restrict__ Q,
                const unsigned short* __restrict__ Kg,
                const unsigned short* __restrict__ V,
                unsigned short* __restrict__ O, int ldq) {
  const int L = 2048, E = 1024, D = 64;
  __shared__ __align__(16) unsigned short Ks[32 * 72];     // key-major, stride 72
  __shared__ __align__(16) unsigned short VsT[64 * 40];    // d-major, stride 40
  __shared__ __align__(16) unsigned short Ps[4][16 * 32];  // per-wave P tile

  const int tid  = threadIdx.x;
  const int w    = tid >> 5;
  const int lane = tid & 31;
  const int half = lane >> 4;
  const int ln   = lane & 15;
  const int b = blockIdx.z, h = blockIdx.y;
  const int q0 = blockIdx.x * 64 + w * 16;

  // Q fragments (16x64 -> two 16x32 A-frags), direct from global
  AF aq[2];
  const size_t qbase = (size_t)(b * L + q0 + ln) * ldq + h * D;
#pragma unroll
  for (int ks = 0; ks < 2; ks++) {
    const unsigned short* p = Q + qbase + ks * 32 + half * 8;
    aq[ks].q[0] = *(const uint4*)(p);
    aq[ks].q[1] = *(const uint4*)(p + 16);
  }

  CF acc[4];
#pragma unroll
  for (int t = 0; t < 4; t++)
#pragma unroll
    for (int e = 0; e < 8; e++) acc[t].f[e] = 0.f;
  float m_run[8], l_run[8];
#pragma unroll
  for (int r = 0; r < 8; r++) { m_run[r] = -1e30f; l_run[r] = 0.f; }

  // cooperative K/V tile loads: 32 rows x 64 cols, 16 elems/thread
  const int kr = tid >> 2;        // key row 0..31
  const int ds = (tid & 3) * 16;  // d segment

  for (int kb = 0; kb < L; kb += 32) {
    const unsigned short* gk = Kg + (size_t)(b * L + kb + kr) * ldq + h * D + ds;
    const unsigned short* gv = V  + (size_t)(b * L + kb + kr) * ldq + h * D + ds;
    union { uint4 q[2]; unsigned short s[16]; } vu;
    vu.q[0] = *(const uint4*)gv;
    vu.q[1] = *(const uint4*)(gv + 8);
#if !USE_ASYNC_LDS
    uint4 kr0 = *(const uint4*)gk;
    uint4 kr1 = *(const uint4*)(gk + 8);
#endif

    __syncthreads();
#if USE_ASYNC_LDS
    async_b128(gk,     &Ks[kr * 72 + ds]);
    async_b128(gk + 8, &Ks[kr * 72 + ds + 8]);
#else
    *(uint4*)(&Ks[kr * 72 + ds])     = kr0;
    *(uint4*)(&Ks[kr * 72 + ds + 8]) = kr1;
#endif
#pragma unroll
    for (int e = 0; e < 16; e++) VsT[(ds + e) * 40 + kr] = vu.s[e];  // transpose
    wait_async0();
    __syncthreads();

    // S = Q @ K^T : two 16x16 tiles (keys 0-15 / 16-31), K-dim = 64 (2 steps)
    CF s0, s1;
#pragma unroll
    for (int e = 0; e < 8; e++) { s0.f[e] = 0.f; s1.f[e] = 0.f; }
#pragma unroll
    for (int t = 0; t < 2; t++) {
#pragma unroll
      for (int ks = 0; ks < 2; ks++) {
        BF bk;  // lane = key column; K = d index, contiguous in Ks row
        const unsigned short* p = &Ks[(t * 16 + ln) * 72 + ks * 32 + half * 8];
        bk.q[0] = *(const uint4*)(p);
        bk.q[1] = *(const uint4*)(p + 16);
        if (t == 0)
          s0.v = __builtin_amdgcn_wmma_f32_16x16x32_bf16(false, aq[ks].v, false, bk.v,
                                                         (short)0, s0.v, false, false);
        else
          s1.v = __builtin_amdgcn_wmma_f32_16x16x32_bf16(false, aq[ks].v, false, bk.v,
                                                         (short)0, s1.v, false, false);
      }
    }

    // online softmax; row (r + 8*half) lives in VGPR r, cols across 16 lanes
#pragma unroll
    for (int r = 0; r < 8; r++) {
      float x0 = s0.f[r] * 0.03125f;
      float x1 = s1.f[r] * 0.03125f;
      float t = red_max16(fmaxf(x0, x1));
      const float mnew  = fmaxf(m_run[r], t);
      const float alpha = __expf(m_run[r] - mnew);
      const float p0 = __expf(x0 - mnew);
      const float p1 = __expf(x1 - mnew);
      float rs = red_sum16(p0 + p1);
      m_run[r] = mnew;
      l_run[r] = l_run[r] * alpha + rs;
#pragma unroll
      for (int t4 = 0; t4 < 4; t4++) acc[t4].f[r] *= alpha;
      const int mrow = r + half * 8;
      Ps[w][mrow * 32 + ln]      = f2bf(p0);
      Ps[w][mrow * 32 + 16 + ln] = f2bf(p1);
    }

    // re-layout P (C layout -> A fragment) through per-wave LDS scratch
    AF ap;
    const unsigned short* pp = &Ps[w][ln * 32 + half * 8];
    ap.q[0] = *(const uint4*)(pp);
    ap.q[1] = *(const uint4*)(pp + 16);

    // O += P @ V : four 16x16 d-tiles
#pragma unroll
    for (int t4 = 0; t4 < 4; t4++) {
      BF bv;  // lane = d column; K = key index, contiguous in VsT row
      const unsigned short* p = &VsT[(t4 * 16 + ln) * 40 + half * 8];
      bv.d[0] = *(const uint2*)(p);
      bv.d[1] = *(const uint2*)(p + 4);
      bv.d[2] = *(const uint2*)(p + 16);
      bv.d[3] = *(const uint2*)(p + 20);
      acc[t4].v = __builtin_amdgcn_wmma_f32_16x16x32_bf16(false, ap.v, false, bv.v,
                                                          (short)0, acc[t4].v, false, false);
    }
  }

  // finalize: O /= l, write bf16 concat-head layout [B*L, H*D]
#pragma unroll
  for (int r = 0; r < 8; r++) {
    const float inv = 1.f / l_run[r];
    const size_t row = (size_t)(b * L + q0 + r + half * 8) * E + h * D;
#pragma unroll
    for (int t4 = 0; t4 < 4; t4++)
      O[row + t4 * 16 + ln] = f2bf(acc[t4].f[r] * inv);
  }
}

// ---------------------------------------------------------------------------
// launch
// ---------------------------------------------------------------------------
extern "C" void kernel_launch(void* const* d_in, const int* in_sizes, int n_in,
                              void* d_out, int out_size, void* d_ws, size_t ws_size,
                              hipStream_t stream) {
  const float* x  = (const float*)d_in[0];
  const float* Wq = (const float*)d_in[1];
  const float* Wk = (const float*)d_in[2];
  const float* Wv = (const float*)d_in[3];
  const float* Wo = (const float*)d_in[4];
  const float* bo = (const float*)d_in[5];

  const int B = 4, L = 2048, E = 1024, H = 16;
  const int EQ = 3 * E;                 // fused QKV width
  const size_t nx = (size_t)B * L * E;  // 8,388,608
  const size_t nw = (size_t)E * E;      // 1,048,576

  unsigned short* ws   = (unsigned short*)d_ws;
  unsigned short* xb   = ws;                         // nx
  unsigned short* Wcat = xb + nx;                    // 1024 x 3072
  unsigned short* Wob  = Wcat + 3 * nw;              // 1024 x 1024
  unsigned short* QKV  = Wob + nw;                   // (B*L) x 3072
  unsigned short* Ab   = QKV + (size_t)B * L * EQ;   // (B*L) x 1024

  cvt4_f32_bf16<<<1024, 256, 0, stream>>>(x, xb, (int)(nx / 4));
  cvt4_w_cat<<<256, 256, 0, stream>>>(Wq, Wcat + 0,    (int)(nw / 4));
  cvt4_w_cat<<<256, 256, 0, stream>>>(Wk, Wcat + 1024, (int)(nw / 4));
  cvt4_w_cat<<<256, 256, 0, stream>>>(Wv, Wcat + 2048, (int)(nw / 4));
  cvt4_f32_bf16<<<256, 256, 0, stream>>>(Wo, Wob, (int)(nw / 4));

  // QKV = x @ [Wq|Wk|Wv]  : M=8192, N=3072, K=1024
  gemm_bf16<0><<<dim3(EQ / 128, (B * L) / 128), 256, 0, stream>>>(
      xb, E, Wcat, EQ, QKV, EQ, nullptr, E);

  attn_flash<<<dim3(L / 64, H, B), 128, 0, stream>>>(
      QKV, QKV + 1024, QKV + 2048, Ab, EQ);

  // out = attn @ Wo + bo : M=8192, N=1024, K=1024, fp32 out
  gemm_bf16<1><<<dim3(E / 128, (B * L) / 128), 256, 0, stream>>>(
      Ab, E, Wob, E, d_out, E, bo, E);
}
